// ChunkedLayer_70205535420729
// MI455X (gfx1250) — matrix operand order repaired and checked by reading, not compiled, not run on hardware
//
#include <hip/hip_runtime.h>
#include <hip/hip_bf16.h>

// ---------------------------------------------------------------------------
// Types for CDNA5 WMMA (gfx1250, wave32)
// ---------------------------------------------------------------------------
typedef __bf16 bf16_t;
typedef __attribute__((ext_vector_type(16))) __bf16 v16bf;
typedef __attribute__((ext_vector_type(8)))  __bf16 v8bf;
typedef __attribute__((ext_vector_type(8)))  float  v8f;

union ABfrag { v16bf v; v8bf h[2]; };

#define Bsz 2
#define Lsz 1024
#define Dsz 1024
#define Hsz 16
#define DKsz 64
#define DVsz 64
#define Msz (Bsz * Lsz)        // 2048 token rows

__device__ __forceinline__ float dev_sigmoid(float x) { return 1.0f / (1.0f + __expf(-x)); }
__device__ __forceinline__ float dev_silu(float x)    { return x / (1.0f + __expf(-x)); }

__device__ __forceinline__ v8f wmma_bf16(const ABfrag& a, const ABfrag& b, v8f c) {
    return __builtin_amdgcn_wmma_f32_16x16x32_bf16(
        /*neg_a=*/false, a.v, /*neg_b=*/false, b.v,
        /*c_mod=*/(short)0, c, /*reuse_a=*/false, /*reuse_b=*/false);
}

// Epilogue: C/D layout — VGPR r -> row mt+r (lanes 0-15) / mt+8+r (lanes 16-31),
// col nt+(lane&15). mode: 0 = silu for n<silu_cols else identity; 1 = sigmoid;
// 2 = identity.
__device__ __forceinline__ void store_tile(float* __restrict__ C, v8f acc,
                                           int mt, int nt, int N,
                                           const float* __restrict__ bias,
                                           int mode, int silu_cols, int lane) {
    const int n = nt + (lane & 15);
    const float bn = bias ? bias[n] : 0.0f;
    const int mbase = mt + ((lane >> 4) << 3);
#pragma unroll
    for (int r = 0; r < 8; ++r) {
        float x = acc[r] + bn;
        if (mode == 0)      { if (n < silu_cols) x = dev_silu(x); }
        else if (mode == 1) { x = dev_sigmoid(x); }
        C[(size_t)(mbase + r) * N + n] = x;
    }
}

// ---------------------------------------------------------------------------
// Pack kernels: fp32 -> bf16, weights transposed to [N x K] row-major so the
// WMMA B fragment is two contiguous 8-element (16B) loads per lane, 16 k apart.
// ---------------------------------------------------------------------------
__global__ void pack_hs_kernel(const float* __restrict__ hs, bf16_t* __restrict__ hsb, int n) {
    int i = blockIdx.x * blockDim.x + threadIdx.x;
    if (i < n) hsb[i] = (bf16_t)hs[i];
}

__global__ void pack_wqkvg_kernel(const float* __restrict__ Wq, const float* __restrict__ Wk,
                                  const float* __restrict__ Wv, const float* __restrict__ Wg,
                                  const float* __restrict__ bq, const float* __restrict__ bk,
                                  const float* __restrict__ bv, const float* __restrict__ bg,
                                  bf16_t* __restrict__ Wt, float* __restrict__ biasA) {
    int idx = blockIdx.x * blockDim.x + threadIdx.x;       // 4096*1024 elements
    if (idx >= 4096 * 1024) return;
    int n = idx >> 10, k = idx & 1023;
    int sel = n >> 10, c = n & 1023;
    const float* W = (sel == 0) ? Wq : (sel == 1) ? Wk : (sel == 2) ? Wv : Wg;
    Wt[(size_t)n * 1024 + k] = (bf16_t)W[(size_t)k * 1024 + c];
    if (k == 0) {
        const float* bb_ = (sel == 0) ? bq : (sel == 1) ? bk : (sel == 2) ? bv : bg;
        biasA[n] = bb_[c];
    }
}

__global__ void pack_wsc_kernel(const float* __restrict__ Wb,  const float* __restrict__ bb,
                                const float* __restrict__ Wfd, const float* __restrict__ bfd,
                                const float* __restrict__ fdb,
                                const float* __restrict__ Wsd, const float* __restrict__ bsd,
                                const float* __restrict__ sdb,
                                const float* __restrict__ Wfg, const float* __restrict__ bfg,
                                const float* __restrict__ Wsg, const float* __restrict__ bsg,
                                bf16_t* __restrict__ Wt, float* __restrict__ biasS) {
    int idx = blockIdx.x * blockDim.x + threadIdx.x;       // 80*1024 elements
    if (idx >= 80 * 1024) return;
    int n = idx >> 10, k = idx & 1023;
    int grp = n >> 4, h = n & 15;
    const float* W = (grp == 0) ? Wb : (grp == 1) ? Wfd : (grp == 2) ? Wsd : (grp == 3) ? Wfg : Wsg;
    Wt[(size_t)n * 1024 + k] = (bf16_t)W[(size_t)k * 16 + h];
    if (k == 0) {
        float bias;
        if      (grp == 0) bias = bb[h];
        else if (grp == 1) bias = bfd[h] + fdb[h];
        else if (grp == 2) bias = bsd[h] + sdb[h];
        else if (grp == 3) bias = bfg[h];
        else               bias = bsg[h];
        biasS[n] = bias;
    }
}

__global__ void pack_wo_kernel(const float* __restrict__ Wo, bf16_t* __restrict__ WtO) {
    int idx = blockIdx.x * blockDim.x + threadIdx.x;       // 1024*1024 elements
    if (idx >= 1024 * 1024) return;
    int n = idx >> 10, k = idx & 1023;
    WtO[(size_t)n * 1024 + k] = (bf16_t)Wo[(size_t)k * 1024 + n];
}

// ---------------------------------------------------------------------------
// Main WMMA bf16 GEMM: each wave computes a 32x32 output tile (2x2 wmma grid,
// 2 A-frags x 2 B-frags reused -> 2 VMEM loads per v_wmma instead of 4).
// Block = 8 waves (2 M x 4 N) -> 64x128 block tile. Requires M%64==0, N%128==0.
// ---------------------------------------------------------------------------
__global__ __launch_bounds__(256) void gemm_bf16_32x32_kernel(
        const bf16_t* __restrict__ A, const bf16_t* __restrict__ Bt,
        const float* __restrict__ bias, float* __restrict__ C,
        int M, int N, int K, int mode, int silu_cols) {
    const int lane = threadIdx.x & 31;
    const int wave = threadIdx.x >> 5;
    const int waveM = wave & 1, waveN = wave >> 1;
    const int m0 = blockIdx.y * 64 + waveM * 32;
    const int n0 = blockIdx.x * 128 + waveN * 32;
    const int l15 = lane & 15;
    const int kAoff = (lane < 16) ? 0 : 8;    // A: lanes 0-15 hold K 0..7 & 16..23
    const int kBoff = (lane < 16) ? 0 : 8;    // B: lanes 0-15 hold K 0..7 & 16..23

    const bf16_t* __restrict__ arow0 = A  + (size_t)(m0 + l15) * K;
    const bf16_t* __restrict__ arow1 = arow0 + (size_t)16 * K;
    const bf16_t* __restrict__ brow0 = Bt + (size_t)(n0 + l15) * K;
    const bf16_t* __restrict__ brow1 = brow0 + (size_t)16 * K;

    v8f acc00 = {}, acc01 = {}, acc10 = {}, acc11 = {};
    for (int k0 = 0; k0 < K; k0 += 32) {
        ABfrag a0, a1, b0, b1;
        a0.h[0] = *(const v8bf*)(arow0 + k0 + kAoff);
        a0.h[1] = *(const v8bf*)(arow0 + k0 + kAoff + 16);
        a1.h[0] = *(const v8bf*)(arow1 + k0 + kAoff);
        a1.h[1] = *(const v8bf*)(arow1 + k0 + kAoff + 16);
        b0.h[0] = *(const v8bf*)(brow0 + k0 + kBoff);
        b0.h[1] = *(const v8bf*)(brow0 + k0 + kBoff + 16);
        b1.h[0] = *(const v8bf*)(brow1 + k0 + kBoff);
        b1.h[1] = *(const v8bf*)(brow1 + k0 + kBoff + 16);
        acc00 = wmma_bf16(a0, b0, acc00);
        acc01 = wmma_bf16(a0, b1, acc01);
        acc10 = wmma_bf16(a1, b0, acc10);
        acc11 = wmma_bf16(a1, b1, acc11);
    }
    store_tile(C, acc00, m0,      n0,      N, bias, mode, silu_cols, lane);
    store_tile(C, acc01, m0,      n0 + 16, N, bias, mode, silu_cols, lane);
    store_tile(C, acc10, m0 + 16, n0,      N, bias, mode, silu_cols, lane);
    store_tile(C, acc11, m0 + 16, n0 + 16, N, bias, mode, silu_cols, lane);
}

// ---------------------------------------------------------------------------
// Narrow-N WMMA GEMM (16x16 tile per wave) for the N=80 scalar-head pass.
// Block = 8 waves -> 128x16 block tile.
// ---------------------------------------------------------------------------
__global__ __launch_bounds__(256) void gemm_bf16_16x16_kernel(
        const bf16_t* __restrict__ A, const bf16_t* __restrict__ Bt,
        const float* __restrict__ bias, float* __restrict__ C,
        int M, int N, int K, int mode, int silu_cols) {
    const int lane = threadIdx.x & 31;
    const int wave = threadIdx.x >> 5;
    const int n0 = blockIdx.x * 16;
    const int m0 = blockIdx.y * 128 + wave * 16;
    const int kAoff = (lane < 16) ? 0 : 8;
    const int kBoff = (lane < 16) ? 0 : 8;

    const bf16_t* __restrict__ arow = A  + (size_t)(m0 + (lane & 15)) * K;
    const bf16_t* __restrict__ brow = Bt + (size_t)(n0 + (lane & 15)) * K;

    v8f acc = {};
    for (int k0 = 0; k0 < K; k0 += 32) {
        ABfrag a, b;
        a.h[0] = *(const v8bf*)(arow + k0 + kAoff);
        a.h[1] = *(const v8bf*)(arow + k0 + kAoff + 16);
        b.h[0] = *(const v8bf*)(brow + k0 + kBoff);
        b.h[1] = *(const v8bf*)(brow + k0 + kBoff + 16);
        acc = wmma_bf16(a, b, acc);
    }
    store_tile(C, acc, m0, n0, N, bias, mode, silu_cols, lane);
}

// ---------------------------------------------------------------------------
// Sequential two-state delta-rule scan. One block per (b,h) chain; 256 threads:
// thread (g = tid>>6, v = tid&63) owns rows g*16..g*16+15 of column v of both
// 64x64 states, held in registers. LDS used for per-step broadcasts and
// partial reductions; wave-level shfl for Frobenius-norm partials.
// ---------------------------------------------------------------------------
__global__ __launch_bounds__(256) void scan_kernel(
        const float* __restrict__ QKVG,   // [2048 x 4096]: q|k|v|g
        const float* __restrict__ SC,     // [2048 x 80]: beta|fd|sd|fg|sg
        const float* __restrict__ rf_p,   // scalar resonance_flux
        float* __restrict__ O) {          // [2048 x 1024]
    const int bh = blockIdx.x;            // 0..31
    const int b = bh >> 4, h = bh & 15;
    const int t = threadIdx.x;
    const int v = t & 63;
    const int g = t >> 6;
    const int lane = t & 31;
    const int wave = t >> 5;
    const float rf = rf_p[0];

    __shared__ float sh_q[64], sh_k[64], sh_v[64];
    __shared__ float sh_scal[8];                 // beta fd sd fg sg rnq rnk
    __shared__ float part_f[4][64], part_s[4][64];
    __shared__ float err_f[64], err_s[64];
    __shared__ float wsum_f[8], wsum_s[8];
    __shared__ float sc_f, sc_s;

    float Sf[16], Ss[16];
#pragma unroll
    for (int i = 0; i < 16; ++i) { Sf[i] = 0.0f; Ss[i] = 0.0f; }

    for (int step = 0; step < Lsz; ++step) {
        const int tok = b * Lsz + step;
        const float* row = QKVG + (size_t)tok * 4096;
        if (t < 64) {
            sh_q[t] = row[h * 64 + t];
            sh_k[t] = row[1024 + h * 64 + t];
        } else if (t < 128) {
            sh_v[t - 64] = row[2048 + h * 64 + (t - 64)];
        } else if (t < 133) {
            sh_scal[t - 128] = SC[(size_t)tok * 80 + (t - 128) * 16 + h];
        }
        __syncthreads();

        // L2 norms of q (wave 0) and k (wave 1)
        if (wave < 2) {
            const float* src = (wave == 0) ? sh_q : sh_k;
            float x0 = src[lane], x1 = src[lane + 32];
            float s = x0 * x0 + x1 * x1;
            for (int off = 16; off > 0; off >>= 1) s += __shfl_down(s, off, 32);
            if (lane == 0) sh_scal[5 + wave] = rsqrtf(s + 1e-6f);
        }
        __syncthreads();

        const float beta = sh_scal[0], fd = sh_scal[1], sd = sh_scal[2];
        const float fg = sh_scal[3], sg = sh_scal[4];
        const float rnq = sh_scal[5], rnk = sh_scal[6];

        // decay + k.S partials
        float kv[16];
        float pf = 0.0f, ps = 0.0f;
#pragma unroll
        for (int i = 0; i < 16; ++i) {
            kv[i] = sh_k[g * 16 + i] * rnk;
            Sf[i] *= fd;
            Ss[i] *= sd;
            pf += kv[i] * Sf[i];
            ps += kv[i] * Ss[i];
        }
        part_f[g][v] = pf; part_s[g][v] = ps;
        __syncthreads();

        if (t < 64) {
            float vv = sh_v[t];
            err_f[t] = vv - (part_f[0][t] + part_f[1][t] + part_f[2][t] + part_f[3][t]);
            err_s[t] = vv - (part_s[0][t] + part_s[1][t] + part_s[2][t] + part_s[3][t]);
        }
        __syncthreads();

        // delta update + resonance coupling; local Frobenius partials
        const float ef = err_f[v], es = err_s[v];
        float nf[16], ns[16];
        float lf = 0.0f, ls = 0.0f;
#pragma unroll
        for (int i = 0; i < 16; ++i) {
            float bk = beta * kv[i];
            float sf = Sf[i] + bk * ef;
            float ss = Ss[i] + bk * es;
            nf[i] = sf + rf * ss;
            ns[i] = ss + rf * sf;
            lf += nf[i] * nf[i];
            ls += ns[i] * ns[i];
        }
        for (int off = 16; off > 0; off >>= 1) {
            lf += __shfl_down(lf, off, 32);
            ls += __shfl_down(ls, off, 32);
        }
        if (lane == 0) { wsum_f[wave] = lf; wsum_s[wave] = ls; }
        __syncthreads();

        if (t == 0) {
            float tf = 0.0f, ts = 0.0f;
            for (int w = 0; w < 8; ++w) { tf += wsum_f[w]; ts += wsum_s[w]; }
            sc_f = 1.0f / fmaxf(1.0f, sqrtf(tf) / (float)DKsz);
            sc_s = 1.0f / fmaxf(1.0f, sqrtf(ts) / (float)DKsz);
        }
        __syncthreads();

        // cap states, q.S partials
        const float cf = sc_f, cs = sc_s;
        float of = 0.0f, osum = 0.0f;
#pragma unroll
        for (int i = 0; i < 16; ++i) {
            Sf[i] = nf[i] * cf;
            Ss[i] = ns[i] * cs;
            float qv = sh_q[g * 16 + i] * rnq;
            of += qv * Sf[i];
            osum += qv * Ss[i];
        }
        part_f[g][v] = of; part_s[g][v] = osum;
        __syncthreads();

        if (t < 64) {
            float o = fg * (part_f[0][t] + part_f[1][t] + part_f[2][t] + part_f[3][t])
                    + sg * (part_s[0][t] + part_s[1][t] + part_s[2][t] + part_s[3][t]);
            O[(size_t)tok * 1024 + h * 64 + t] = o;
        }
        // No trailing barrier needed: next-iter LDS writes are fenced by the
        // load barrier, and part arrays are rewritten only after two barriers.
    }
}

// ---------------------------------------------------------------------------
// Gated RMSNorm epilogue: one wave32 per (token, head); lanes cover DV=64 as 2.
// Produces bf16 activations for the final WMMA output projection.
// ---------------------------------------------------------------------------
__global__ __launch_bounds__(256) void normgate_kernel(
        const float* __restrict__ O, const float* __restrict__ QKVG,
        const float* __restrict__ onorm_w, bf16_t* __restrict__ A2) {
    int grp = blockIdx.x * 8 + (threadIdx.x >> 5);   // token*16 + h
    int lane = threadIdx.x & 31;
    int tok = grp >> 4, h = grp & 15;
    const float* o  = O    + (size_t)tok * 1024 + h * 64;
    const float* gg = QKVG + (size_t)tok * 4096 + 3072 + h * 64;
    float x0 = o[lane], x1 = o[lane + 32];
    float s = x0 * x0 + x1 * x1;
    for (int off = 16; off > 0; off >>= 1) s += __shfl_down(s, off, 32);
    s = __shfl(s, 0, 32);
    float r = rsqrtf(s / (float)DVsz + 1e-5f);
    float y0 = x0 * r * onorm_w[lane]      * dev_silu(gg[lane]);
    float y1 = x1 * r * onorm_w[lane + 32] * dev_silu(gg[lane + 32]);
    A2[(size_t)tok * 1024 + h * 64 + lane]      = (bf16_t)y0;
    A2[(size_t)tok * 1024 + h * 64 + lane + 32] = (bf16_t)y1;
}

// ---------------------------------------------------------------------------
// Host-side launch
// ---------------------------------------------------------------------------
extern "C" void kernel_launch(void* const* d_in, const int* in_sizes, int n_in,
                              void* d_out, int out_size, void* d_ws, size_t ws_size,
                              hipStream_t stream) {
    const float* hs   = (const float*)d_in[0];
    const float* Wq   = (const float*)d_in[1];  const float* bq  = (const float*)d_in[2];
    const float* Wk   = (const float*)d_in[3];  const float* bk  = (const float*)d_in[4];
    const float* Wv   = (const float*)d_in[5];  const float* bv  = (const float*)d_in[6];
    const float* Wb   = (const float*)d_in[7];  const float* bb  = (const float*)d_in[8];
    const float* Wfd  = (const float*)d_in[9];  const float* bfd = (const float*)d_in[10];
    const float* fdb  = (const float*)d_in[11];
    const float* Wsd  = (const float*)d_in[12]; const float* bsd = (const float*)d_in[13];
    const float* sdb  = (const float*)d_in[14];
    const float* Wfg  = (const float*)d_in[15]; const float* bfg = (const float*)d_in[16];
    const float* Wsg  = (const float*)d_in[17]; const float* bsg = (const float*)d_in[18];
    const float* Wg   = (const float*)d_in[19]; const float* bg  = (const float*)d_in[20];
    const float* onw  = (const float*)d_in[21];
    const float* Wo   = (const float*)d_in[22]; const float* bo  = (const float*)d_in[23];
    const float* rf   = (const float*)d_in[24];
    float* out = (float*)d_out;

    // Workspace carve-up (256B aligned)
    char* base = (char*)d_ws;
    size_t off = 0;
    auto carve = [&](size_t bytes) -> char* {
        char* p = base + off;
        off = (off + bytes + 255) & ~(size_t)255;
        return p;
    };
    bf16_t* hsb   = (bf16_t*)carve((size_t)Msz * Dsz * 2);         // 4 MiB
    bf16_t* WtA   = (bf16_t*)carve((size_t)4096 * Dsz * 2);        // 8 MiB
    float*  biasA = (float*) carve((size_t)4096 * 4);
    bf16_t* WtS   = (bf16_t*)carve((size_t)80 * Dsz * 2);
    float*  biasS = (float*) carve((size_t)80 * 4);
    bf16_t* WtO   = (bf16_t*)carve((size_t)Dsz * Dsz * 2);         // 2 MiB
    float*  QKVG  = (float*) carve((size_t)Msz * 4096 * 4);        // 32 MiB
    float*  SC    = (float*) carve((size_t)Msz * 80 * 4);
    float*  O     = (float*) carve((size_t)Msz * Dsz * 4);         // 8 MiB
    bf16_t* A2    = (bf16_t*)carve((size_t)Msz * Dsz * 2);         // 4 MiB
    (void)ws_size;

    // 1) Packs (bf16 conversion + transpose + bias fusion)
    pack_hs_kernel<<<(Msz * Dsz + 255) / 256, 256, 0, stream>>>(hs, hsb, Msz * Dsz);
    pack_wqkvg_kernel<<<(4096 * 1024 + 255) / 256, 256, 0, stream>>>(
        Wq, Wk, Wv, Wg, bq, bk, bv, bg, WtA, biasA);
    pack_wsc_kernel<<<(80 * 1024 + 255) / 256, 256, 0, stream>>>(
        Wb, bb, Wfd, bfd, fdb, Wsd, bsd, sdb, Wfg, bfg, Wsg, bsg, WtS, biasS);
    pack_wo_kernel<<<(1024 * 1024 + 255) / 256, 256, 0, stream>>>(Wo, WtO);

    // 2) Fused q|k|v|g projection GEMM (silu on q,k,v; raw g), 32x32/wave tiles
    {
        dim3 grid(4096 / 128, Msz / 64);
        gemm_bf16_32x32_kernel<<<grid, 256, 0, stream>>>(hsb, WtA, biasA, QKVG,
                                                         Msz, 4096, Dsz, /*mode=*/0,
                                                         /*silu_cols=*/3072);
    }
    // 3) Scalar-head GEMM (sigmoid): beta|fd|sd|fg|sg  (N=80 -> 16-wide tiles)
    {
        dim3 grid(80 / 16, Msz / 128);
        gemm_bf16_16x16_kernel<<<grid, 256, 0, stream>>>(hsb, WtS, biasS, SC,
                                                         Msz, 80, Dsz, /*mode=*/1, 0);
    }
    // 4) Sequential two-state scan: one block per (b,h)
    scan_kernel<<<Bsz * Hsz, 256, 0, stream>>>(QKVG, SC, rf, O);

    // 5) Gated RMSNorm -> bf16
    normgate_kernel<<<(Msz * Hsz) / 8, 256, 0, stream>>>(O, QKVG, onw, A2);

    // 6) Output projection GEMM -> d_out (f32), 32x32/wave tiles
    {
        dim3 grid(Dsz / 128, Msz / 64);
        gemm_bf16_32x32_kernel<<<grid, 256, 0, stream>>>(A2, WtO, bo, out,
                                                         Msz, Dsz, Dsz, /*mode=*/2, 0);
    }
    (void)in_sizes; (void)n_in; (void)out_size;
}
